// SelfCorrelationComputation3_584115552555
// MI455X (gfx1250) — matrix operand order repaired and checked by reading, not compiled, not run on hardware
//
#include <hip/hip_runtime.h>
#include <hip/hip_bf16.h>
#include <math.h>

// ---------------------------------------------------------------------------
// Non-local block (self-attention) for MI455X / gfx1250, wave32 + WMMA bf16.
// All GEMMs use v_wmma_f32_16x16x32_bf16 (fp32 accum). Layout conventions per
// CDNA5 ISA 7.12.2 for C/D; A/B use this code's contiguous k order, not the ISA table:
//   A (16x32 bf16): lane L holds row M=L&15; half h=L>>4: elems j -> K=k0+16h+j (contig).
//   B (32x16 bf16): lane L holds col N=L&15; half h: elems j -> K=k0+16h+j (contig).
//   C/D (16x16 f32): lane L holds col N=L&15; vgpr v -> row M=v+8*(L>>4).
// ---------------------------------------------------------------------------

typedef __attribute__((ext_vector_type(16))) __bf16 v16bf;
typedef __attribute__((ext_vector_type(8)))  __bf16 v8bf;
typedef __attribute__((ext_vector_type(8)))  float  v8f;

#define NB    16
#define C_IN  256
#define CI    128
#define WW    64
#define NSP   4096   // H*W
#define NK    1024   // pooled spatial (32*32)
#define BN_EPS 1e-5f

static __device__ inline v8f vzero() {
  v8f z;
#pragma unroll
  for (int i = 0; i < 8; ++i) z[i] = 0.f;
  return z;
}

static __device__ inline v8f wmma_bf16(v16bf a, v16bf b, v8f c) {
  return __builtin_amdgcn_wmma_f32_16x16x32_bf16(false, a, false, b, (short)0, c,
                                                 false, false);
}

// A-operand load: rowPtr points at element [row, 0] of a K-contiguous row.
static __device__ inline v16bf loadA_bf16(const __bf16* rowPtr, int k0, int half) {
  const v8bf lo = *reinterpret_cast<const v8bf*>(rowPtr + k0 + half * 16);
  const v8bf hi = *reinterpret_cast<const v8bf*>(rowPtr + k0 + half * 16 + 8);
  v16bf a;
#pragma unroll
  for (int i = 0; i < 8; ++i) { a[i] = lo[i]; a[i + 8] = hi[i]; }
  return a;
}

// A-operand load from f32 source with bf16 conversion.
static __device__ inline v16bf loadA_f32cvt(const float* rowPtr, int k0, int half) {
  v16bf a;
#pragma unroll
  for (int j = 0; j < 8; ++j) {
    a[j]     = (__bf16)rowPtr[k0 + half * 16 + j];
    a[j + 8] = (__bf16)rowPtr[k0 + half * 16 + 8 + j];
  }
  return a;
}

// B-operand load: colPtr points at element [col, 0] of a K-contiguous "row"
// of the (transposed) storage; 16 contiguous K values per lane.
static __device__ inline v16bf loadB_bf16(const __bf16* colPtr, int k0, int half) {
  const v8bf lo = *reinterpret_cast<const v8bf*>(colPtr + k0 + half * 16);
  const v8bf hi = *reinterpret_cast<const v8bf*>(colPtr + k0 + half * 16 + 8);
  v16bf b;
#pragma unroll
  for (int i = 0; i < 8; ++i) { b[i] = lo[i]; b[i + 8] = hi[i]; }
  return b;
}

static __device__ inline float half_reduce_max(float v) {
  v = fmaxf(v, __shfl_xor(v, 1, 16));
  v = fmaxf(v, __shfl_xor(v, 2, 16));
  v = fmaxf(v, __shfl_xor(v, 4, 16));
  v = fmaxf(v, __shfl_xor(v, 8, 16));
  return v;
}

static __device__ inline float half_reduce_sum(float v) {
  v += __shfl_xor(v, 1, 16);
  v += __shfl_xor(v, 2, 16);
  v += __shfl_xor(v, 4, 16);
  v += __shfl_xor(v, 8, 16);
  return v;
}

// ---------------------------------------------------------------------------
// Kernel 1: the three 1x1 convs as GEMMs, FUSED: one wave computes the same
// 16(ci) x 16(n) tile for theta, g and phi, sharing the x B-operand tile
// (3 WMMAs per loaded B tile -> 3x less x/L2 traffic than one-conv-per-wave).
// theta -> [B][N][CI] bf16 ; g_full/phi_full -> [B][CI][N] bf16 (for pooling)
// ---------------------------------------------------------------------------
__global__ __launch_bounds__(256)
void k1_conv1x1(const float* __restrict__ x,
                const float* __restrict__ g_w,  const float* __restrict__ g_b,
                const float* __restrict__ th_w, const float* __restrict__ th_b,
                const float* __restrict__ ph_w, const float* __restrict__ ph_b,
                __bf16* __restrict__ theta,
                __bf16* __restrict__ g_full,
                __bf16* __restrict__ phi_full) {
  const int lane = threadIdx.x & 31;
  const int half = lane >> 4;
  const int ln15 = lane & 15;

  int wid = blockIdx.x * 8 + (threadIdx.x >> 5);
  const int nt = wid & 255; wid >>= 8;        // spatial tile 0..255
  const int ct = wid & 7;   wid >>= 3;        // ci tile 0..7
  const int b = wid;                          // batch 0..15

  const int ciBase = ct * 16, nBase = nt * 16;
  const int n = nBase + ln15;                           // D/B column (spatial)
  const size_t wro = (size_t)(ciBase + ln15) * C_IN;    // A row offset (ci)
  const float* xcol = x + (size_t)b * C_IN * NSP + n;

  v8f accT = vzero(), accG = vzero(), accP = vzero();
#pragma unroll
  for (int kc = 0; kc < C_IN / 32; ++kc) {
    const int k0 = kc * 32;
    // B from x: lane = spatial col, K = input channel (strided by NSP)
    v16bf bm;
#pragma unroll
    for (int j = 0; j < 16; ++j)
      bm[j] = (__bf16)xcol[(size_t)(k0 + half * 16 + j) * NSP];
    // three A tiles against the shared B tile
    accT = wmma_bf16(loadA_f32cvt(th_w + wro, k0, half), bm, accT);
    accG = wmma_bf16(loadA_f32cvt(g_w  + wro, k0, half), bm, accG);
    accP = wmma_bf16(loadA_f32cvt(ph_w + wro, k0, half), bm, accP);
  }

  // bias per output row ci = ciBase + 8*half + v  (contiguous over v)
  const v8f bT = *reinterpret_cast<const v8f*>(th_b + ciBase + 8 * half);
  const v8f bG = *reinterpret_cast<const v8f*>(g_b  + ciBase + 8 * half);
  const v8f bP = *reinterpret_cast<const v8f*>(ph_b + ciBase + 8 * half);

  // theta: [B][N][CI] -> one contiguous 16B store per lane
  v8bf st;
#pragma unroll
  for (int v = 0; v < 8; ++v) st[v] = (__bf16)(accT[v] + bT[v]);
  *reinterpret_cast<v8bf*>(theta + ((size_t)b * NSP + n) * CI + ciBase + 8 * half) = st;

  // g/phi: [B][CI][N] channel-major for the pooling pass
#pragma unroll
  for (int v = 0; v < 8; ++v) {
    const int ci = ciBase + 8 * half + v;
    const size_t o = ((size_t)b * CI + ci) * NSP + n;
    g_full[o]   = (__bf16)(accG[v] + bG[v]);
    phi_full[o] = (__bf16)(accP[v] + bP[v]);
  }
}

// ---------------------------------------------------------------------------
// Kernel 1b: 2x2 maxpool. Produces g_pool [B][CI][NK] (Nk contiguous) and
// phi_pool [B][NK][CI] (ci contiguous) -> contiguous WMMA B-operand loads.
// ---------------------------------------------------------------------------
__global__ __launch_bounds__(256)
void k1b_pool(const __bf16* __restrict__ g_full, const __bf16* __restrict__ phi_full,
              __bf16* __restrict__ g_pool, __bf16* __restrict__ phi_pool) {
  int idx = blockIdx.x * 256 + threadIdx.x;   // B*CI*NK threads
  const int nk = idx & (NK - 1); idx >>= 10;
  const int ci = idx & (CI - 1); idx >>= 7;
  const int b  = idx;
  const int hk = nk >> 5, wk = nk & 31;
  const size_t base = ((size_t)b * CI + ci) * NSP + (size_t)(hk * 2) * WW + wk * 2;

  float m0 = fmaxf((float)g_full[base],        (float)g_full[base + 1]);
  float m1 = fmaxf((float)g_full[base + WW],   (float)g_full[base + WW + 1]);
  g_pool[((size_t)b * CI + ci) * NK + nk] = (__bf16)fmaxf(m0, m1);

  float p0 = fmaxf((float)phi_full[base],      (float)phi_full[base + 1]);
  float p1 = fmaxf((float)phi_full[base + WW], (float)phi_full[base + WW + 1]);
  phi_pool[((size_t)b * NK + nk) * CI + ci] = (__bf16)fmaxf(p0, p1);
}

// ---------------------------------------------------------------------------
// Kernel 2: flash-attention. One wave owns 16 theta rows; iterates Nk in
// chunks of 32 with online softmax. 16 WMMAs per chunk. Prefetches the next
// phi chunk (global_prefetch_b8) to overlap L2 fetch with WMMA/softmax work.
// ---------------------------------------------------------------------------
__global__ __launch_bounds__(128)
void k2_attention(const __bf16* __restrict__ theta,  // [B][N][CI]
                  const __bf16* __restrict__ phi,    // [B][NK][CI]
                  const __bf16* __restrict__ g,      // [B][CI][NK]
                  __bf16* __restrict__ y) {          // [B][N][CI]
  __shared__ __bf16 pls[4][16 * 32];                 // per-wave P transpose tile
  const int lane = threadIdx.x & 31;
  const int half = lane >> 4, ln15 = lane & 15;
  const int wslot = threadIdx.x >> 5;
  const int wid = blockIdx.x * 4 + wslot;
  const int tile = wid & 255;
  const int b = wid >> 8;
  const int rowBase = tile * 16;

  // Preload theta A-tiles for all of K=CI (4 chunks of 32).
  const __bf16* arow = theta + ((size_t)b * NSP + rowBase + ln15) * CI;
  v16bf A[4];
#pragma unroll
  for (int kc = 0; kc < 4; ++kc) A[kc] = loadA_bf16(arow, kc * 32, half);

  float mrun[8], srun[8];
  v8f acc[8];
#pragma unroll
  for (int v = 0; v < 8; ++v) { mrun[v] = -INFINITY; srun[v] = 0.f; }
#pragma unroll
  for (int t = 0; t < 8; ++t) acc[t] = vzero();

  const __bf16* gbase = g + (size_t)b * CI * NK;

  for (int nkc = 0; nkc < NK / 32; ++nkc) {
    const int nk0 = nkc * 32;
    const __bf16* bcol0 = phi + ((size_t)b * NK + nk0 + ln15) * CI;
    const __bf16* bcol1 = phi + ((size_t)b * NK + nk0 + 16 + ln15) * CI;

    // prefetch next phi chunk while this one computes (uniform branch)
    if (nkc + 1 < NK / 32) {
      __builtin_prefetch(bcol0 + 32 * CI, 0, 3);
      __builtin_prefetch(bcol1 + 32 * CI, 0, 3);
    }

    // --- logits: f[16 x 32] = theta_tile x phi_chunk (8 WMMAs) ---
    v8f f0 = vzero(), f1 = vzero();
#pragma unroll
    for (int kc = 0; kc < 4; ++kc) {
      f0 = wmma_bf16(A[kc], loadB_bf16(bcol0, kc * 32, half), f0);
      f1 = wmma_bf16(A[kc], loadB_bf16(bcol1, kc * 32, half), f1);
    }
    // --- online softmax: stats per row m = v + 8*half ---
    float sc[8];
#pragma unroll
    for (int v = 0; v < 8; ++v) {
      float rm = half_reduce_max(fmaxf(f0[v], f1[v]));
      float nm = fmaxf(mrun[v], rm);
      sc[v] = __expf(mrun[v] - nm);
      mrun[v] = nm;
    }
#pragma unroll
    for (int v = 0; v < 8; ++v) {
      float p0 = __expf(f0[v] - mrun[v]);
      float p1 = __expf(f1[v] - mrun[v]);
      srun[v] = srun[v] * sc[v] + half_reduce_sum(p0 + p1);
      const int r = v + 8 * half;
      pls[wslot][r * 32 + ln15]      = (__bf16)p0;   // D-layout -> LDS (row-major)
      pls[wslot][r * 32 + 16 + ln15] = (__bf16)p1;
    }
#pragma unroll
    for (int t = 0; t < 8; ++t)
#pragma unroll
      for (int v = 0; v < 8; ++v) acc[t][v] *= sc[v];

    // wave-local LDS transpose: wait for our DS stores, reload in A layout
    asm volatile("s_wait_dscnt 0" ::: "memory");
    v16bf PA = loadA_bf16(&pls[wslot][ln15 * 32], 0, half);

    // --- y += P x g_chunk : 8 WMMAs over CI=128 columns ---
#pragma unroll
    for (int t = 0; t < 8; ++t) {
      v16bf Bg = loadB_bf16(gbase + (size_t)(t * 16 + ln15) * NK, nk0, half);
      acc[t] = wmma_bf16(PA, Bg, acc[t]);
    }
  }

  // normalize and store y[B][N][CI]
  float inv[8];
#pragma unroll
  for (int v = 0; v < 8; ++v) inv[v] = 1.f / srun[v];
#pragma unroll
  for (int t = 0; t < 8; ++t)
#pragma unroll
    for (int v = 0; v < 8; ++v) {
      const int n = rowBase + v + 8 * half;
      y[((size_t)b * NSP + n) * CI + t * 16 + ln15] = (__bf16)(acc[t][v] * inv[v]);
    }
}

// ---------------------------------------------------------------------------
// Kernel 3: out = BN(W_w x y + W_b) + x. M=C_IN, N=spatial, K=CI. One wave per
// 16x16 tile, 4 WMMAs, fused bias/BN/residual epilogue.
// ---------------------------------------------------------------------------
__global__ __launch_bounds__(256)
void k3_out(const __bf16* __restrict__ y,   // [B][N][CI]
            const float* __restrict__ Ww,   // [C][CI]
            const float* __restrict__ Wb,
            const float* __restrict__ gamma, const float* __restrict__ beta,
            const float* __restrict__ mean,  const float* __restrict__ var,
            const float* __restrict__ x,
            float* __restrict__ out) {
  const int lane = threadIdx.x & 31;
  const int half = lane >> 4, ln15 = lane & 15;
  int wid = blockIdx.x * 8 + (threadIdx.x >> 5);
  const int nt = wid & 255; wid >>= 8;
  const int ot = wid & 15;  wid >>= 4;
  const int b = wid;
  const int oBase = ot * 16, nBase = nt * 16;

  const float* wrow = Ww + (size_t)(oBase + ln15) * CI;
  const __bf16* ycol = y + ((size_t)b * NSP + nBase + ln15) * CI;

  v8f acc = vzero();
#pragma unroll
  for (int kc = 0; kc < CI / 32; ++kc) {
    const int k0 = kc * 32;
    acc = wmma_bf16(loadA_f32cvt(wrow, k0, half), loadB_bf16(ycol, k0, half), acc);
  }

  const int o8 = oBase + 8 * half;  // rows o = o8 + v, contiguous over v
  const v8f vb  = *reinterpret_cast<const v8f*>(Wb + o8);
  const v8f vg  = *reinterpret_cast<const v8f*>(gamma + o8);
  const v8f vbe = *reinterpret_cast<const v8f*>(beta + o8);
  const v8f vm  = *reinterpret_cast<const v8f*>(mean + o8);
  const v8f vv  = *reinterpret_cast<const v8f*>(var + o8);
  const int n = nBase + ln15;
#pragma unroll
  for (int v = 0; v < 8; ++v) {
    const int o = o8 + v;
    float val = acc[v] + vb[v];
    val = (val - vm[v]) * (vg[v] * rsqrtf(vv[v] + BN_EPS)) + vbe[v];
    const size_t idx = ((size_t)b * C_IN + o) * NSP + n;
    out[idx] = val + x[idx];
  }
}

// ---------------------------------------------------------------------------
extern "C" void kernel_launch(void* const* d_in, const int* in_sizes, int n_in,
                              void* d_out, int out_size, void* d_ws, size_t ws_size,
                              hipStream_t stream) {
  (void)in_sizes; (void)n_in; (void)out_size; (void)ws_size;
  const float* x    = (const float*)d_in[0];
  const float* g_w  = (const float*)d_in[1];
  const float* g_b  = (const float*)d_in[2];
  const float* th_w = (const float*)d_in[3];
  const float* th_b = (const float*)d_in[4];
  const float* ph_w = (const float*)d_in[5];
  const float* ph_b = (const float*)d_in[6];
  const float* W_w  = (const float*)d_in[7];
  const float* W_b  = (const float*)d_in[8];
  const float* bn_g = (const float*)d_in[9];
  const float* bn_b = (const float*)d_in[10];
  const float* bn_m = (const float*)d_in[11];
  const float* bn_v = (const float*)d_in[12];
  float* out = (float*)d_out;

  // Workspace carve-up (56 MiB total):
  char* ws = (char*)d_ws;
  const size_t SZ_FULL = (size_t)NB * CI * NSP * sizeof(__bf16);   // 16 MiB
  const size_t SZ_POOL = (size_t)NB * CI * NK * sizeof(__bf16);    //  4 MiB
  __bf16* theta    = (__bf16*)(ws);
  __bf16* g_full   = (__bf16*)(ws + SZ_FULL);
  __bf16* phi_full = (__bf16*)(ws + 2 * SZ_FULL);
  __bf16* g_pool   = (__bf16*)(ws + 3 * SZ_FULL);
  __bf16* phi_pool = (__bf16*)(ws + 3 * SZ_FULL + SZ_POOL);
  __bf16* ybuf     = g_full;  // g_full dead after pooling; reuse for y

  // K1: B*(CI/16)*(NSP/16) = 32768 wave-tiles (3 convs fused), 8 waves/block
  k1_conv1x1<<<4096, 256, 0, stream>>>(x, g_w, g_b, th_w, th_b, ph_w, ph_b,
                                       theta, g_full, phi_full);
  // K1b: B*CI*NK = 2097152 threads
  k1b_pool<<<8192, 256, 0, stream>>>(g_full, phi_full, g_pool, phi_pool);
  // K2: B*(NSP/16) = 4096 wave-tiles, 4 waves/block
  k2_attention<<<1024, 128, 0, stream>>>(theta, phi_pool, g_pool, ybuf);
  // K3: B*(C/16)*(NSP/16) = 65536 wave-tiles, 8 waves/block
  k3_out<<<8192, 256, 0, stream>>>(ybuf, W_w, W_b, bn_g, bn_b, bn_m, bn_v, x, out);
}
